// AttentionMap_42013370089933
// MI455X (gfx1250) — matrix operand order repaired and checked by reading, not compiled, not run on hardware
//
#include <hip/hip_runtime.h>
#include <hip/hip_bf16.h>

// ---------------------------------------------------------------------------
// Causal attention forward for MI455X (gfx1250, wave32, WMMA).
//   out  = softmax(QK^T/8 + causal_bias) @ V      [B,H,S,D]
//   wts  = softmax(QK^T/8 + causal_bias)          [B,H,S,S]
// d_out = [out | wts] flat f32.
//
// Roofline: mandatory weights write = 537 MB (~24 us @ 23.3 TB/s); dense
// FLOPs ~17-34 GFLOP -> must run on v_wmma_f32_16x16x32_f16 (f32 WMMA is
// ~8x weaker per instruction) to stay bandwidth-bound. Two-pass (recompute)
// flash softmax writes the weights exactly once. Softmax runs in base-2
// domain so exp lowers to a single v_exp_f32 per element.
// ---------------------------------------------------------------------------

typedef __attribute__((ext_vector_type(16))) _Float16 v16h;
typedef __attribute__((ext_vector_type(8)))  _Float16 v8h;
typedef __attribute__((ext_vector_type(8)))  float    v8f;

#define WMMA_F16F32(a, b, c)                                                   \
  __builtin_amdgcn_wmma_f32_16x16x32_f16(false, (a), false, (b), (short)0,     \
                                         (c), false, false)

namespace {
constexpr int kB = 2;
constexpr int kH = 16;
constexpr int kS = 2048;
constexpr int kD = 64;
constexpr float kLog2e   = 1.4426950408889634f;
constexpr float kScale2  = 0.125f * kLog2e;     // (1/sqrt(64)) * log2(e)
constexpr float kNegB2   = -1e9f * kLog2e;      // causal bias in base-2 domain
}  // namespace

__device__ __forceinline__ float fast_exp2(float x) {
  return __builtin_amdgcn_exp2f(x);             // v_exp_f32
}

// One 16(q) x 16(k) score tile in base-2 domain:
//   s2 = (Q K^T) * scale * log2e + causal_bias2
// a_lo/a_hi: Q in WMMA A-layout (d 0..31 / 32..63). Returns C-layout v8f:
//   lane L: n = L&15, rows M = r + 8*(L>>4).
__device__ __forceinline__ v8f score_tile(const float* __restrict__ Kb,
                                          int k0, int qt, int kt, int lane,
                                          const v16h& a_lo, const v16h& a_hi) {
  const int half = lane >> 4;
  const int l16  = lane & 15;
  // B operand in this code's contiguous k order (same as A): lane<16 -> column n=l16, Kdim 0..15;
  //                        lane>=16 -> column n=l16, Kdim 16..31.
  const float* kr = Kb + (size_t)(k0 + l16) * kD + half * 16;
  v16h b_lo, b_hi;
#pragma unroll
  for (int c = 0; c < 4; ++c) {
    float4 f0 = *(const float4*)(kr + 4 * c);         // d = half*16 + 4c ..
    float4 f1 = *(const float4*)(kr + 32 + 4 * c);    // d = 32 + half*16 + 4c
    b_lo[4 * c + 0] = (_Float16)f0.x; b_lo[4 * c + 1] = (_Float16)f0.y;
    b_lo[4 * c + 2] = (_Float16)f0.z; b_lo[4 * c + 3] = (_Float16)f0.w;
    b_hi[4 * c + 0] = (_Float16)f1.x; b_hi[4 * c + 1] = (_Float16)f1.y;
    b_hi[4 * c + 2] = (_Float16)f1.z; b_hi[4 * c + 3] = (_Float16)f1.w;
  }
  v8f acc = {};
  acc = WMMA_F16F32(a_lo, b_lo, acc);   // d = 0..31
  acc = WMMA_F16F32(a_hi, b_hi, acc);   // d = 32..63
  v8f s;
#pragma unroll
  for (int r = 0; r < 8; ++r) {
    float v = acc[r] * kScale2;
    const int M = r + 8 * half;             // row within q tile
    if (kt == qt && l16 > M) v += kNegB2;   // causal: key > query
    s[r] = v;
  }
  return s;
}

__global__ __launch_bounds__(32) void attn_fwd_kernel(
    const float* __restrict__ Q, const float* __restrict__ K,
    const float* __restrict__ V, float* __restrict__ Out,
    float* __restrict__ W) {
  const int qt   = blockIdx.x;   // 0..127 query tile
  const int bh   = blockIdx.y;   // 0..31  (b*H + h)
  const int lane = threadIdx.x;  // 0..31
  const int half = lane >> 4;
  const int l16  = lane & 15;
  const int q0   = qt * 16;

  __shared__ _Float16 pbuf[16 * 32];   // P tile, row-major [M][kk]
  __shared__ _Float16 vbuf[64 * 32];   // V^T tile, [d][key_rel]

  const float* Qb = Q + (size_t)bh * kS * kD;
  const float* Kb = K + (size_t)bh * kS * kD;
  const float* Vb = V + (size_t)bh * kS * kD;

  // ---- Q tile into WMMA A-layout (16x32 halves x2), kept for both passes.
  // contiguous k order (same as B): lane<16: row l16, K = 0..15; lane>=16: row l16, K = 16..31
  const int koff = half * 16;
  const float* qrow = Qb + (size_t)(q0 + l16) * kD;
  v16h a_lo, a_hi;
  {
    float4 f0 = *(const float4*)(qrow + koff);
    float4 f1 = *(const float4*)(qrow + koff + 4);
    float4 f2 = *(const float4*)(qrow + koff + 8);
    float4 f3 = *(const float4*)(qrow + koff + 12);
    float4 f4 = *(const float4*)(qrow + 32 + koff);
    float4 f5 = *(const float4*)(qrow + 32 + koff + 4);
    float4 f6 = *(const float4*)(qrow + 32 + koff + 8);
    float4 f7 = *(const float4*)(qrow + 32 + koff + 12);
    a_lo[0]=(_Float16)f0.x; a_lo[1]=(_Float16)f0.y; a_lo[2]=(_Float16)f0.z; a_lo[3]=(_Float16)f0.w;
    a_lo[4]=(_Float16)f1.x; a_lo[5]=(_Float16)f1.y; a_lo[6]=(_Float16)f1.z; a_lo[7]=(_Float16)f1.w;
    a_lo[8]=(_Float16)f2.x; a_lo[9]=(_Float16)f2.y; a_lo[10]=(_Float16)f2.z; a_lo[11]=(_Float16)f2.w;
    a_lo[12]=(_Float16)f3.x; a_lo[13]=(_Float16)f3.y; a_lo[14]=(_Float16)f3.z; a_lo[15]=(_Float16)f3.w;
    a_hi[0]=(_Float16)f4.x; a_hi[1]=(_Float16)f4.y; a_hi[2]=(_Float16)f4.z; a_hi[3]=(_Float16)f4.w;
    a_hi[4]=(_Float16)f5.x; a_hi[5]=(_Float16)f5.y; a_hi[6]=(_Float16)f5.z; a_hi[7]=(_Float16)f5.w;
    a_hi[8]=(_Float16)f6.x; a_hi[9]=(_Float16)f6.y; a_hi[10]=(_Float16)f6.z; a_hi[11]=(_Float16)f6.w;
    a_hi[12]=(_Float16)f7.x; a_hi[13]=(_Float16)f7.y; a_hi[14]=(_Float16)f7.z; a_hi[15]=(_Float16)f7.w;
  }

  // ====== Pass A: per-lane online softmax stats (base-2 domain) ======
  float m[8], l[8];
#pragma unroll
  for (int r = 0; r < 8; ++r) { m[r] = -INFINITY; l[r] = 0.0f; }

  for (int kt = 0; kt <= qt; ++kt) {
    // Prefetch next K tile (clamped so the address stays inside the tensor).
    {
      const int ktn = (kt < qt) ? kt + 1 : kt;
      __builtin_prefetch(Kb + (size_t)(ktn * 16 + l16) * kD + half * 16, 0, 0);
    }
    v8f s = score_tile(Kb, kt * 16, qt, kt, lane, a_lo, a_hi);
#pragma unroll
    for (int r = 0; r < 8; ++r) {
      float mo = m[r];
      float mn = fmaxf(mo, s[r]);
      l[r] = l[r] * fast_exp2(mo - mn) + fast_exp2(s[r] - mn);
      m[r] = mn;
    }
  }
  // Combine (m,l) across the 16-lane group (each row lives in 16 lanes).
#pragma unroll
  for (int off = 1; off < 16; off <<= 1) {
#pragma unroll
    for (int r = 0; r < 8; ++r) {
      float om = __shfl_xor(m[r], off, 32);
      float ol = __shfl_xor(l[r], off, 32);
      float mn = fmaxf(m[r], om);
      l[r] = l[r] * fast_exp2(m[r] - mn) + ol * fast_exp2(om - mn);
      m[r] = mn;
    }
  }
  float rl[8];
#pragma unroll
  for (int r = 0; r < 8; ++r) rl[r] = __builtin_amdgcn_rcpf(l[r]);

  // ================= Pass B: write P, accumulate O = P @ V =================
  v8f o[4];
#pragma unroll
  for (int j = 0; j < 4; ++j) o[j] = {};

  for (int ktp = 0; ktp <= qt; ktp += 2) {
    // --- score tile ktp (and ktp+1 if in range): normalize, store, pack ---
#pragma unroll
    for (int t = 0; t < 2; ++t) {
      const int kt = ktp + t;
      if (kt <= qt) {
        v8f s = score_tile(Kb, kt * 16, qt, kt, lane, a_lo, a_hi);
#pragma unroll
        for (int r = 0; r < 8; ++r) {
          const int M = r + 8 * half;
          float p = fast_exp2(s[r] - m[r]) * rl[r];  // masked -> exp2(-1e9..)=0
          W[(size_t)(bh * kS + q0 + M) * kS + kt * 16 + l16] = p;
          pbuf[M * 32 + t * 16 + l16] = (_Float16)p;
        }
      } else {
        // zero-pad second K-half of the P tile (odd number of k tiles)
#pragma unroll
        for (int r = 0; r < 8; ++r)
          pbuf[(r + 8 * half) * 32 + 16 + l16] = (_Float16)0.0f;
      }
    }

    // --- stage V^T for keys [k0p, k0p+32): vbuf[d][key_rel] as f16.
    // Lane owns a KEY PAIR (2*l16, 2*l16+1) over half the d-range so each
    // LDS store packs two f16 into one b32 (v_cvt_pk_f16_f32 feeds it).
    {
      const int   dbase = 32 * half;                        // 0 or 32
      const float* r0 = Vb + (size_t)(ktp * 16 + 2 * l16) * kD + dbase;
      const float* r1 = r0 + kD;                            // odd key of pair
#pragma unroll
      for (int dd = 0; dd < 32; dd += 4) {
        float4 fa = *(const float4*)(r0 + dd);
        float4 fb = *(const float4*)(r1 + dd);
        union { _Float16 h[2]; unsigned u; } p0, p1, p2, p3;
        p0.h[0] = (_Float16)fa.x; p0.h[1] = (_Float16)fb.x;
        p1.h[0] = (_Float16)fa.y; p1.h[1] = (_Float16)fb.y;
        p2.h[0] = (_Float16)fa.z; p2.h[1] = (_Float16)fb.z;
        p3.h[0] = (_Float16)fa.w; p3.h[1] = (_Float16)fb.w;
        *(unsigned*)&vbuf[(dbase + dd + 0) * 32 + 2 * l16] = p0.u;
        *(unsigned*)&vbuf[(dbase + dd + 1) * 32 + 2 * l16] = p1.u;
        *(unsigned*)&vbuf[(dbase + dd + 2) * 32 + 2 * l16] = p2.u;
        *(unsigned*)&vbuf[(dbase + dd + 3) * 32 + 2 * l16] = p3.u;
      }
    }
    __syncthreads();

    // --- P (16x32) from LDS in A-layout ---
    v16h p_a;
    {
      const v8h plo = *(const v8h*)&pbuf[l16 * 32 + koff];
      const v8h phi = *(const v8h*)&pbuf[l16 * 32 + 8 + koff];
#pragma unroll
      for (int e = 0; e < 8; ++e) { p_a[e] = plo[e]; p_a[8 + e] = phi[e]; }
    }
    // --- O[:, 16j:16j+16] += P @ V  (B-layout from vbuf) ---
#pragma unroll
    for (int j = 0; j < 4; ++j) {
      const _Float16* vp = &vbuf[(j * 16 + l16) * 32 + half * 16];
      const v8h vlo = *(const v8h*)vp;
      const v8h vhi = *(const v8h*)(vp + 8);
      v16h b_v;
#pragma unroll
      for (int e = 0; e < 8; ++e) { b_v[e] = vlo[e]; b_v[8 + e] = vhi[e]; }
      o[j] = WMMA_F16F32(p_a, b_v, o[j]);
    }
    __syncthreads();
  }

  // ---- store O (C-layout: N across lanes, M across the 8 slots) ----
#pragma unroll
  for (int j = 0; j < 4; ++j) {
#pragma unroll
    for (int r = 0; r < 8; ++r) {
      const int M = r + 8 * half;
      Out[(size_t)(bh * kS + q0 + M) * kD + j * 16 + l16] = o[j][r];
    }
  }

  // ---- zero-fill the strict upper triangle of this q tile's weight rows ----
  const int zc0 = (qt + 1) * 16;
  if (zc0 < kS) {
    const float4 z4 = make_float4(0.f, 0.f, 0.f, 0.f);
    for (int rr = 0; rr < 16; ++rr) {
      float* wr = W + (size_t)(bh * kS + q0 + rr) * kS;
      for (int c = zc0 + lane * 4; c < kS; c += 128)
        *(float4*)(wr + c) = z4;
    }
  }
}

extern "C" void kernel_launch(void* const* d_in, const int* in_sizes, int n_in,
                              void* d_out, int out_size, void* d_ws,
                              size_t ws_size, hipStream_t stream) {
  (void)in_sizes; (void)n_in; (void)d_ws; (void)ws_size; (void)out_size;
  const float* Q = (const float*)d_in[0];
  const float* K = (const float*)d_in[1];
  const float* V = (const float*)d_in[2];
  // d_in[3] = causal mask; causality is applied analytically in-kernel.
  float* Out = (float*)d_out;                       // [B,H,S,D]
  float* W   = Out + (size_t)kB * kH * kS * kD;     // [B,H,S,S]
  dim3 grid(kS / 16, kB * kH);
  attn_fwd_kernel<<<grid, dim3(32), 0, stream>>>(Q, K, V, Out, W);
}
